// GritTransformerLayer_90512140796688
// MI455X (gfx1250) — matrix operand order repaired and checked by reading, not compiled, not run on hardware
//
#include <hip/hip_runtime.h>

// ---------------------------------------------------------------------------
// GRIT transformer layer for MI455X (gfx1250, wave32, WMMA).
// All dense GEMMs use v_wmma_f32_16x16x32_bf16 (bf16 inputs, fp32 accum).
// Weights are pre-converted to bf16 once (L2-resident), so B fragments are
// raw 32B vector loads; only the streaming A operand is converted on the fly.
// ---------------------------------------------------------------------------

typedef __attribute__((ext_vector_type(16))) __bf16 v16bf;
typedef __attribute__((ext_vector_type(8)))  __bf16 v8bf;
typedef __attribute__((ext_vector_type(8)))  float  v8f;

#define N_NODES 30000
#define N_EDGES 480000
#define DIM     128
#define NH      8
#define CLAMPV  5.0f

__device__ __forceinline__ v8f wmma_bf16(v16bf a, v16bf b, v8f c) {
  // D = A(16x32 bf16) * B(32x16 bf16) + C(16x16 f32)
  return __builtin_amdgcn_wmma_f32_16x16x32_bf16(false, a, false, b,
                                                 (short)0, c, false, false);
}

// A fragment: 16x32 bf16. Lane L: row L&15, K = kb..kb+15
// with kb = k0 + 16*(L>=16). Source is fp32 row-major.
__device__ __forceinline__ v16bf load_afrag_f32(const float* __restrict__ p) {
  v16bf a;
#pragma unroll
  for (int i = 0; i < 8; ++i) a[i] = (__bf16)p[i];
#pragma unroll
  for (int i = 0; i < 8; ++i) a[8 + i] = (__bf16)p[8 + i];
  return a;
}
__device__ __forceinline__ v16bf load_afrag_bf16(const __bf16* __restrict__ p) {
  v16bf a;
  v8bf lo = *(const v8bf*)p;
  v8bf hi = *(const v8bf*)(p + 8);
#pragma unroll
  for (int i = 0; i < 8; ++i) { a[i] = lo[i]; a[8 + i] = hi[i]; }
  return a;
}
// B fragment: 32x16 bf16, B[k,n] = W[n,k]. Lane L holds column n = L&15,
// K = kb..kb+15 with kb = k0 + 16*(L>=16): 16 contiguous bf16 of row n of Wb.
__device__ __forceinline__ v16bf load_bfrag_bf16(const __bf16* __restrict__ p) {
  return *(const v16bf*)p;  // 32B aligned by construction
}

__device__ __forceinline__ void atomicMaxFloat(float* addr, float val) {
  if (val >= 0.0f) atomicMax((int*)addr, __float_as_int(val));
  else             atomicMin((unsigned int*)addr, __float_as_uint(val));
}

// ---------------------------------------------------------------------------
// fp32 -> bf16 weight pre-conversion (runs once per launch; tiny)
// ---------------------------------------------------------------------------
__global__ void cvt_bf16_kernel(const float* __restrict__ src,
                                __bf16* __restrict__ dst, int nElem) {
  for (int id = blockIdx.x * blockDim.x + threadIdx.x; id < nElem;
       id += gridDim.x * blockDim.x)
    dst[id] = (__bf16)src[id];
}

// ---------------------------------------------------------------------------
// Generic Y[M,outN] = act(X[M,inK] @ Wb[outN,inK]^T + bias (+ residual))
// One wave per 16-row tile. inK = KSTEPS*32 (templated: A frags stay in regs).
// ---------------------------------------------------------------------------
template <int KSTEPS>
__global__ __launch_bounds__(256) void gemm_xwT_kernel(
    const float* __restrict__ X, const __bf16* __restrict__ Wb,
    const float* __restrict__ bias, const float* __restrict__ residual,
    float* __restrict__ Y, int M, int outN, int doRelu) {
  const int inK  = KSTEPS * 32;
  const int gid  = blockIdx.x * blockDim.x + threadIdx.x;
  const int wave = gid >> 5;
  const int lane = threadIdx.x & 31;
  const int tiles = M >> 4;
  if (wave >= tiles) return;
  const int row0 = wave << 4;

  v16bf afr[KSTEPS];
  {
    const int r  = row0 + (lane & 15);
    const int ko = (lane >> 4) << 4;  // 0 or 16
#pragma unroll
    for (int kk = 0; kk < KSTEPS; ++kk)
      afr[kk] = load_afrag_f32(X + (size_t)r * inK + kk * 32 + ko);
  }
  const int n   = lane & 15;
  const int mb  = (lane >> 4) << 3;  // D tile: m = mb + v
  const int kob = (lane >> 4) << 4;  // B frag K offset: 0 or 16

  for (int j = 0; j < (outN >> 4); ++j) {
    v8f acc = {};
    const int col = (j << 4) + n;
    const __bf16* wrow = Wb + (size_t)col * inK;
#pragma unroll
    for (int kk = 0; kk < KSTEPS; ++kk)
      acc = wmma_bf16(afr[kk], load_bfrag_bf16(wrow + kk * 32 + kob), acc);
    const float bv = bias ? bias[col] : 0.0f;
#pragma unroll
    for (int v = 0; v < 8; ++v) {
      const int rr = row0 + mb + v;
      float val = acc[v] + bv;
      if (residual) val += residual[(size_t)rr * outN + col];
      if (doRelu)   val = fmaxf(val, 0.0f);
      Y[(size_t)rr * outN + col] = val;
    }
  }
}

// A is bf16 (edge_w), inK = 128 fixed, residual required.
__global__ __launch_bounds__(256) void gemm_bf16A_kernel(
    const __bf16* __restrict__ Xb, const __bf16* __restrict__ Wb,
    const float* __restrict__ bias, const float* __restrict__ residual,
    float* __restrict__ Y, int M) {
  const int gid  = blockIdx.x * blockDim.x + threadIdx.x;
  const int wave = gid >> 5;
  const int lane = threadIdx.x & 31;
  const int tiles = M >> 4;
  if (wave >= tiles) return;
  const int row0 = wave << 4;

  v16bf afr[4];
  {
    const int r  = row0 + (lane & 15);
    const int ko = (lane >> 4) << 4;
#pragma unroll
    for (int kk = 0; kk < 4; ++kk)
      afr[kk] = load_afrag_bf16(Xb + (size_t)r * 128 + kk * 32 + ko);
  }
  const int n = lane & 15, mb = (lane >> 4) << 3, kob = (lane >> 4) << 4;
  for (int j = 0; j < 8; ++j) {
    v8f acc = {};
    const int col = (j << 4) + n;
    const __bf16* wrow = Wb + (size_t)col * 128;
#pragma unroll
    for (int kk = 0; kk < 4; ++kk)
      acc = wmma_bf16(afr[kk], load_bfrag_bf16(wrow + kk * 32 + kob), acc);
    const float bv = bias[col];
#pragma unroll
    for (int v = 0; v < 8; ++v) {
      const int rr = row0 + mb + v;
      Y[(size_t)rr * 128 + col] =
          acc[v] + bv + residual[(size_t)rr * 128 + col];
    }
  }
}

// ---------------------------------------------------------------------------
// Fused edge kernel: E = edge_attr@We^T+be (WMMA -> LDS, never HBM), then
// edge_w = relu((Kh[src]+Qh[dst])*Ew+Eb), score = clip(edge_w . Aw),
// atomic segment-max of score over dst. 4 waves/block, 16 edges/wave.
// ---------------------------------------------------------------------------
__global__ __launch_bounds__(128) void edge_phase1_kernel(
    const float* __restrict__ edge_attr, const __bf16* __restrict__ Web,
    const float* __restrict__ be, const int* __restrict__ ei,
    const float* __restrict__ Qh, const float* __restrict__ Kh,
    const float* __restrict__ Aw, __bf16* __restrict__ edge_w,
    float* __restrict__ score, float* __restrict__ smax) {
  __shared__ float Elds[4][16][256];  // 64 KB
  const int w    = threadIdx.x >> 5;
  const int lane = threadIdx.x & 31;
  const int tile = blockIdx.x * 4 + w;
  const int e0   = tile << 4;

  // --- GEMM phase: 16x256 tile of E into LDS ---
  v16bf afr[4];
  {
    const int r  = e0 + (lane & 15);
    const int ko = (lane >> 4) << 4;
#pragma unroll
    for (int kk = 0; kk < 4; ++kk)
      afr[kk] = load_afrag_f32(edge_attr + (size_t)r * 128 + kk * 32 + ko);
  }
  const int n = lane & 15, mb = (lane >> 4) << 3, kob = (lane >> 4) << 4;
  for (int j = 0; j < 16; ++j) {
    v8f acc = {};
    const int col = (j << 4) + n;
    const __bf16* wrow = Web + (size_t)col * 128;
#pragma unroll
    for (int kk = 0; kk < 4; ++kk)
      acc = wmma_bf16(afr[kk], load_bfrag_bf16(wrow + kk * 32 + kob), acc);
    const float bv = be[col];
#pragma unroll
    for (int v = 0; v < 8; ++v) Elds[w][mb + v][col] = acc[v] + bv;
  }
  __syncthreads();

  // --- elementwise phase: 16 edges x 8 heads = 128 pairs, 4 per lane ---
#pragma unroll
  for (int p = 0; p < 4; ++p) {
    const int q = lane + (p << 5);  // 0..127
    const int e = q & 15, h = q >> 4;
    const int eg  = e0 + e;
    const int src = ei[eg];
    const int dst = ei[N_EDGES + eg];
    const float* kr = Kh + (size_t)src * 128 + h * 16;
    const float* qr = Qh + (size_t)dst * 128 + h * 16;
    float ew[16];
    float sc = 0.0f;
#pragma unroll
    for (int d = 0; d < 16; ++d) {
      float v = (kr[d] + qr[d]) * Elds[w][e][h * 16 + d] +
                Elds[w][e][128 + h * 16 + d];
      v = fmaxf(v, 0.0f);
      ew[d] = v;
      sc += v * Aw[d * NH + h];
    }
    v8bf p0, p1;
#pragma unroll
    for (int d = 0; d < 8; ++d) { p0[d] = (__bf16)ew[d]; p1[d] = (__bf16)ew[8 + d]; }
    __bf16* dstp = edge_w + (size_t)eg * 128 + h * 16;
    *(v8bf*)dstp = p0;
    *(v8bf*)(dstp + 8) = p1;
    sc = fminf(fmaxf(sc, -CLAMPV), CLAMPV);
    score[(size_t)eg * NH + h] = sc;
    atomicMaxFloat(&smax[(size_t)dst * NH + h], sc);
  }
}

// ex = exp(score - smax[dst]) (in place), ssum += ex
__global__ void ex_ssum_kernel(const int* __restrict__ ei,
                               float* __restrict__ scoreio,
                               const float* __restrict__ smax,
                               float* __restrict__ ssum) {
  const size_t total = (size_t)N_EDGES * NH;
  for (size_t id = (size_t)blockIdx.x * blockDim.x + threadIdx.x; id < total;
       id += (size_t)gridDim.x * blockDim.x) {
    const int e = (int)(id >> 3), h = (int)(id & 7);
    const int dst = ei[N_EDGES + e];
    const float v = __expf(scoreio[id] - smax[(size_t)dst * NH + h]);
    scoreio[id] = v;
    atomicAdd(&ssum[(size_t)dst * NH + h], v);
  }
}

// attn = ex/(ssum[dst]+eps); wV += Vh[src]*attn; rowV += edge_w*attn
__global__ void attn_acc_kernel(const int* __restrict__ ei,
                                const float* __restrict__ ex,
                                const float* __restrict__ ssum,
                                const float* __restrict__ Vh,
                                const __bf16* __restrict__ edge_w,
                                float* __restrict__ wV,
                                float* __restrict__ rowV) {
  const size_t total = (size_t)N_EDGES * NH;
  for (size_t id = (size_t)blockIdx.x * blockDim.x + threadIdx.x; id < total;
       id += (size_t)gridDim.x * blockDim.x) {
    const int e = (int)(id >> 3), h = (int)(id & 7);
    const int src = ei[e];
    const int dst = ei[N_EDGES + e];
    const float a = ex[id] / (ssum[(size_t)dst * NH + h] + 1e-16f);
    const float4* vr4 = (const float4*)(Vh + (size_t)src * 128 + h * 16);
    const __bf16* er  = edge_w + (size_t)e * 128 + h * 16;
    const v8bf e0 = *(const v8bf*)er;
    const v8bf e1 = *(const v8bf*)(er + 8);
    float* wv = wV   + (size_t)dst * 128 + h * 16;
    float* rv = rowV + (size_t)dst * 128 + h * 16;
#pragma unroll
    for (int qd = 0; qd < 4; ++qd) {
      const float4 v4 = vr4[qd];
      atomicAdd(&wv[qd * 4 + 0], v4.x * a);
      atomicAdd(&wv[qd * 4 + 1], v4.y * a);
      atomicAdd(&wv[qd * 4 + 2], v4.z * a);
      atomicAdd(&wv[qd * 4 + 3], v4.w * a);
    }
#pragma unroll
    for (int d = 0; d < 8; ++d) {
      atomicAdd(&rv[d],     (float)e0[d] * a);
      atomicAdd(&rv[8 + d], (float)e1[d] * a);
    }
  }
}

// per node: VeRow einsum + wV + degree gating -> hA
__global__ __launch_bounds__(128) void combine_kernel(
    const float* __restrict__ wV, const float* __restrict__ rowV,
    const float* __restrict__ VeRow, const float* __restrict__ deg,
    const float* __restrict__ deg_coef, float* __restrict__ hA) {
  const int nidx = blockIdx.x;
  const int c = threadIdx.x;          // 0..127
  const int h = c >> 4, cc = c & 15;
  float s = 0.0f;
#pragma unroll
  for (int d = 0; d < 16; ++d)
    s += rowV[(size_t)nidx * 128 + h * 16 + d] * VeRow[d * 128 + h * 16 + cc];
  const float t  = wV[(size_t)nidx * 128 + c] + s;
  const float ld = logf(deg[nidx] + 1.0f);
  hA[(size_t)nidx * 128 + c] = t * deg_coef[2 * c] + t * ld * deg_coef[2 * c + 1];
}

// column sums + sumsq (128 cols) for BatchNorm
__global__ __launch_bounds__(128) void colstats_kernel(
    const float* __restrict__ X, float* __restrict__ stats, int rows) {
  const int c = threadIdx.x;
  float s = 0.0f, q = 0.0f;
  for (int r = blockIdx.x; r < rows; r += gridDim.x) {
    __builtin_prefetch(X + ((size_t)r + gridDim.x) * 128 + c, 0, 0);
    const float v = X[(size_t)r * 128 + c];
    s += v; q += v * v;
  }
  atomicAdd(&stats[c], s);
  atomicAdd(&stats[128 + c], q);
}

__global__ void bn_apply_kernel(const float* __restrict__ X,
                                float* __restrict__ Y,
                                const float* __restrict__ stats,
                                const float* __restrict__ g,
                                const float* __restrict__ b, int rows) {
  const size_t total = (size_t)rows * 128;
  const float inv = 1.0f / (float)rows;
  const size_t stride = (size_t)gridDim.x * blockDim.x;
  for (size_t id = (size_t)blockIdx.x * blockDim.x + threadIdx.x; id < total;
       id += stride) {
    __builtin_prefetch(X + id + stride, 0, 0);
    const int c = (int)(id & 127);
    const float mean = stats[c] * inv;
    const float var  = stats[128 + c] * inv - mean * mean;
    Y[id] = (X[id] - mean) * rsqrtf(var + 1e-5f) * g[c] + b[c];
  }
}

__global__ void fill_kernel(float* __restrict__ p, float v, size_t nElem) {
  for (size_t id = (size_t)blockIdx.x * blockDim.x + threadIdx.x; id < nElem;
       id += (size_t)gridDim.x * blockDim.x)
    p[id] = v;
}

// ---------------------------------------------------------------------------
extern "C" void kernel_launch(void* const* d_in, const int* in_sizes, int n_in,
                              void* d_out, int out_size, void* d_ws,
                              size_t ws_size, hipStream_t stream) {
  const float* x         = (const float*)d_in[0];
  const float* edge_attr = (const float*)d_in[1];
  const int*   ei        = (const int*)d_in[2];
  const float* deg       = (const float*)d_in[3];
  const float* Wq  = (const float*)d_in[4];
  const float* bq  = (const float*)d_in[5];
  const float* Wk  = (const float*)d_in[6];
  const float* We  = (const float*)d_in[7];
  const float* be  = (const float*)d_in[8];
  const float* Wv  = (const float*)d_in[9];
  const float* Aw  = (const float*)d_in[10];
  const float* VeRow = (const float*)d_in[11];
  const float* Woh = (const float*)d_in[12];
  const float* boh = (const float*)d_in[13];
  const float* Woe = (const float*)d_in[14];
  const float* boe = (const float*)d_in[15];
  const float* deg_coef = (const float*)d_in[16];
  const float* Wf1 = (const float*)d_in[17];
  const float* bf1 = (const float*)d_in[18];
  const float* Wf2 = (const float*)d_in[19];
  const float* bf2 = (const float*)d_in[20];
  const float* g1h = (const float*)d_in[21];
  const float* b1h = (const float*)d_in[22];
  const float* g1e = (const float*)d_in[23];
  const float* b1e = (const float*)d_in[24];
  const float* g2h = (const float*)d_in[25];
  const float* b2h = (const float*)d_in[26];

  const size_t ND = (size_t)N_NODES * DIM;
  float* ws    = (float*)d_ws;
  float* Qh    = ws;
  float* Kh    = Qh + ND;
  float* Vh    = Kh + ND;
  float* score = Vh + ND;                       // EG*8 (becomes ex)
  float* smax  = score + (size_t)N_EDGES * NH;  // N*8
  float* ssum  = smax + (size_t)N_NODES * NH;   // N*8
  float* wV    = ssum + (size_t)N_NODES * NH;   // N*128
  float* rowV  = wV + ND;                       // N*128
  float* statsE = rowV + ND;                    // 256
  float* stats1 = statsE + 256;                 // 256
  float* stats2 = stats1 + 256;                 // 256
  __bf16* edge_w = (__bf16*)(stats2 + 256);     // EG*128 bf16 (64B aligned)
  // bf16 weight cache (after edge_w; all offsets keep 32B alignment)
  __bf16* Wq_b  = edge_w + (size_t)N_EDGES * 128;
  __bf16* Wk_b  = Wq_b + 16384;
  __bf16* Wv_b  = Wk_b + 16384;
  __bf16* We_b  = Wv_b + 16384;   // 256*128
  __bf16* Woh_b = We_b + 32768;
  __bf16* Woe_b = Woh_b + 16384;
  __bf16* Wf1_b = Woe_b + 16384;  // 256*128
  __bf16* Wf2_b = Wf1_b + 32768;  // 128*256
  // overlays (lifetimes disjoint):
  float* hA = Qh;   // after attn done
  float* hB = Kh;
  float* hC = Vh;
  float* hD = wV;   // N*256 spans wV+rowV (free after combine)
  float* hE = Qh;   // free after Woh GEMM consumed hA

  float* out_h = (float*)d_out;
  float* out_e = out_h + ND;

  // ---- init + weight pre-conversion ----
  fill_kernel<<<512, 256, 0, stream>>>(smax, -CLAMPV, (size_t)N_NODES * NH);
  fill_kernel<<<2048, 256, 0, stream>>>(ssum, 0.0f,
                                        (size_t)N_NODES * NH + 2 * ND);
  fill_kernel<<<1, 256, 0, stream>>>(statsE, 0.0f, 768);
  cvt_bf16_kernel<<<64, 256, 0, stream>>>(Wq,  Wq_b,  16384);
  cvt_bf16_kernel<<<64, 256, 0, stream>>>(Wk,  Wk_b,  16384);
  cvt_bf16_kernel<<<64, 256, 0, stream>>>(Wv,  Wv_b,  16384);
  cvt_bf16_kernel<<<128, 256, 0, stream>>>(We,  We_b,  32768);
  cvt_bf16_kernel<<<64, 256, 0, stream>>>(Woh, Woh_b, 16384);
  cvt_bf16_kernel<<<64, 256, 0, stream>>>(Woe, Woe_b, 16384);
  cvt_bf16_kernel<<<128, 256, 0, stream>>>(Wf1, Wf1_b, 32768);
  cvt_bf16_kernel<<<128, 256, 0, stream>>>(Wf2, Wf2_b, 32768);

  // ---- node Q,K,V GEMMs ----
  const int nodeTiles = N_NODES / 16;                    // 1875
  const int nodeBlocks = (nodeTiles + 7) / 8;            // 235
  gemm_xwT_kernel<4><<<nodeBlocks, 256, 0, stream>>>(x, Wq_b, bq, nullptr, Qh,
                                                     N_NODES, DIM, 0);
  gemm_xwT_kernel<4><<<nodeBlocks, 256, 0, stream>>>(x, Wk_b, nullptr, nullptr,
                                                     Kh, N_NODES, DIM, 0);
  gemm_xwT_kernel<4><<<nodeBlocks, 256, 0, stream>>>(x, Wv_b, nullptr, nullptr,
                                                     Vh, N_NODES, DIM, 0);

  // ---- fused edge phase 1 ----
  edge_phase1_kernel<<<N_EDGES / 64, 128, 0, stream>>>(
      edge_attr, We_b, be, ei, Qh, Kh, Aw, edge_w, score, smax);

  // ---- softmax over segments ----
  ex_ssum_kernel<<<15000, 256, 0, stream>>>(ei, score, smax, ssum);
  attn_acc_kernel<<<15000, 256, 0, stream>>>(ei, score, ssum, Vh, edge_w, wV,
                                             rowV);

  // ---- edge output: e = BN(edge_attr + wE@Woe^T + boe) ----
  const int edgeBlocks = (N_EDGES / 16 + 7) / 8;  // 3750
  gemm_bf16A_kernel<<<edgeBlocks, 256, 0, stream>>>(edge_w, Woe_b, boe,
                                                    edge_attr, out_e, N_EDGES);
  colstats_kernel<<<1024, 128, 0, stream>>>(out_e, statsE, N_EDGES);
  bn_apply_kernel<<<4096, 256, 0, stream>>>(out_e, out_e, statsE, g1e, b1e,
                                            N_EDGES);

  // ---- node output path ----
  combine_kernel<<<N_NODES, 128, 0, stream>>>(wV, rowV, VeRow, deg, deg_coef,
                                              hA);
  gemm_xwT_kernel<4><<<nodeBlocks, 256, 0, stream>>>(hA, Woh_b, boh, x, hB,
                                                     N_NODES, DIM, 0);
  colstats_kernel<<<1024, 128, 0, stream>>>(hB, stats1, N_NODES);
  bn_apply_kernel<<<2048, 256, 0, stream>>>(hB, hC, stats1, g1h, b1h, N_NODES);
  gemm_xwT_kernel<4><<<nodeBlocks, 256, 0, stream>>>(hC, Wf1_b, bf1, nullptr,
                                                     hD, N_NODES, 2 * DIM, 1);
  gemm_xwT_kernel<8><<<nodeBlocks, 256, 0, stream>>>(hD, Wf2_b, bf2, hC, hE,
                                                     N_NODES, DIM, 0);
  colstats_kernel<<<1024, 128, 0, stream>>>(hE, stats2, N_NODES);
  bn_apply_kernel<<<2048, 256, 0, stream>>>(hE, out_h, stats2, g2h, b2h,
                                            N_NODES);

  (void)in_sizes; (void)n_in; (void)out_size; (void)ws_size;
}
